// INSLoss_18820546691648
// MI455X (gfx1250) — matrix operand order repaired and checked by reading, not compiled, not run on hardware
//
#include <hip/hip_runtime.h>
#include <math.h>

typedef __attribute__((ext_vector_type(16))) _Float16 v16h;
typedef __attribute__((ext_vector_type(8)))  float    v8f;

#define NIMG 8
#define NCH  256
#define HH   64
#define WW   64
#define HW   (HH*WW)          // 4096
#define KBOX 16
#define NBOX (NIMG*KBOX)      // 128
#define RED_BLOCKS 1024
#define LDSPITCH 264          // 256 + 8 halfs: 132-dword row pitch -> conflict-free A gathers

// ---- workspace byte offsets ----
#define OFF_LX     0
#define OFF_LY     512
#define OFF_RX     1024
#define OFF_RY     1536
#define OFF_CNT    2048
#define OFF_DEG    2560
#define OFF_FGT    4096        // _Float16 [NBOX*NCH]            = 64 KB
#define OFF_ENERGY 69632       // float    [NBOX*HW]             = 2 MB
#define OFF_EMIN   2166784     // float    [NBOX]
#define OFF_EMAX   2167296     // float    [NBOX]
#define OFF_MASK   2167808     // float    [NIMG*HW]             = 128 KB (16B aligned)
#define OFF_PART   2298880     // float    [RED_BLOCKS*2]        = 8 KB

union HV { v16h v; _Float16 e[16]; };
union FV { v8f  v; float    e[8];  };

// ---------------- K1: box quantization ----------------
__global__ void k1_boxes(const float* __restrict__ bb, int* lx, int* ly, int* rx, int* ry,
                         float* cnt, int* degen) {
    const int t = threadIdx.x;
    if (t < NBOX) {
        const float* b = bb + t * 4;
        // STRIDE=32 is a power of two, so *(1/32) is exact
        int x0 = (int)floorf(b[0] * 0.03125f);
        int y0 = (int)floorf(b[1] * 0.03125f);
        int x1 = (int)floorf(b[2] * 0.03125f);
        int y1 = (int)floorf(b[3] * 0.03125f);
        x0 = min(max(x0, 0), WW - 1); x1 = min(max(x1, 0), WW - 1);
        y0 = min(max(y0, 0), HH - 1); y1 = min(max(y1, 0), HH - 1);
        lx[t] = x0; rx[t] = x1; ly[t] = y0; ry[t] = y1;
        cnt[t] = fmaxf((float)((x1 - x0 + 1) * (y1 - y0 + 1)), 1.0f);
        degen[t] = ((x0 == x1) || (y0 == y1)) ? 1 : 0;
    }
}

// ---------------- K2: region mean-pool of preds_T -> feat_gt (f16) ----------------
__global__ void k2_featgt(const float* __restrict__ T, const int* __restrict__ lx,
                          const int* __restrict__ ly, const int* __restrict__ rx,
                          const int* __restrict__ ry, const float* __restrict__ cnt,
                          _Float16* __restrict__ fgt) {
    const int b = blockIdx.x;                 // 0..127 box id
    const int n = b >> 4;                     // image
    const int c = threadIdx.x;                // channel 0..255
    const float* f = T + ((size_t)(n * NCH + c)) * HW;
    const int x0 = lx[b], x1 = rx[b], y0 = ly[b], y1 = ry[b];
    float s = 0.0f;
    for (int h = y0; h <= y1; ++h) {
        const float* row = f + h * WW;
        for (int w = x0; w <= x1; ++w) s += row[w];
    }
    fgt[(size_t)b * NCH + c] = (_Float16)(s / cnt[b]);
}

// ---------------- K3: energy GEMM via V_WMMA_F32_16X16X32_F16 ----------------
// Per image n: energy[16 boxes x 4096 px] = feat_gt (16x256) @ feat (256x4096)
// One wave per 16-pixel tile; K=256 accumulated as 8 chunks of 32.
__global__ void k3_energy_wmma(const float* __restrict__ T,
                               const _Float16* __restrict__ fgt,
                               float* __restrict__ energy) {
    __shared__ _Float16 fg[KBOX * LDSPITCH];   // 8448 bytes
    const int n  = blockIdx.x >> 5;            // image
    const int tg = blockIdx.x & 31;            // tile group (8 tiles each)
    const int t  = threadIdx.x;

    // Stage feat_gt[n] (16x256 f16) into LDS with padded pitch
    {
        const int box = t >> 4, seg = t & 15;
        const _Float16* src = fgt + ((size_t)(n * KBOX + box)) * NCH + seg * 16;
        _Float16* dst = fg + box * LDSPITCH + seg * 16;
        #pragma unroll
        for (int j = 0; j < 16; ++j) dst[j] = src[j];
    }
    __syncthreads();

    const int wave  = t >> 5;
    const int lane  = t & 31;
    const int tile  = tg * 8 + wave;           // 0..255
    const int col   = lane & 15;               // N index (pixel within tile)
    const int pixel = tile * 16 + col;
    const int hkA   = (lane < 16) ? 0 : 16;    // 16-bit A 16x32 lane layout (contiguous k, matches B)
    const int hkB   = (lane < 16) ? 0 : 16;    // 16-bit B 32x16 lane layout
    const int M     = lane & 15;               // A row (box)
    const float* Tn = T + (size_t)n * NCH * HW;

    v8f acc;
    {
        FV z;
        #pragma unroll
        for (int i = 0; i < 8; ++i) z.e[i] = 0.0f;
        acc = z.v;
    }

    #pragma unroll
    for (int chunk = 0; chunk < 8; ++chunk) {
        const int base = chunk * 32;
        HV a, b;
        // A fragment: element (M, K); halfs idx -> K=hkA+idx
        #pragma unroll
        for (int idx = 0; idx < 8; ++idx)
            a.e[idx] = fg[M * LDSPITCH + base + hkA + idx];
        #pragma unroll
        for (int idx = 8; idx < 16; ++idx)
            a.e[idx] = fg[M * LDSPITCH + base + hkA + idx];
        // B fragment: element (K, N=pixel); halfs idx -> K=hkB+idx; convert f32->f16 on the fly
        #pragma unroll
        for (int idx = 0; idx < 16; ++idx)
            b.e[idx] = (_Float16)Tn[(size_t)(base + hkB + idx) * HW + pixel];
        // (neg_a, A, neg_b, B, c_mod, C, reuse_a, reuse_b)
        acc = __builtin_amdgcn_wmma_f32_16x16x32_f16(false, a.v, false, b.v,
                                                     (short)0, acc, false, false);
    }

    // D layout: lanes 0-15: (M=v, N=lane); lanes 16-31: (M=v+8, N=lane-16)
    FV d; d.v = acc;
    const int rowAdd = (lane < 16) ? 0 : 8;
    #pragma unroll
    for (int v = 0; v < 8; ++v) {
        const int box = v + rowAdd;
        energy[((size_t)(n * KBOX + box)) * HW + tile * 16 + col] = d.e[v];
    }
}

// ---------------- K4: per-box min/max of energy map ----------------
__global__ void k4_minmax(const float* __restrict__ energy,
                          float* __restrict__ emin, float* __restrict__ emax) {
    __shared__ float smin[256], smax[256];
    const int b = blockIdx.x;
    const float* e = energy + (size_t)b * HW;
    float mn = 3.402823e38f, mx = -3.402823e38f;
    for (int i = threadIdx.x; i < HW; i += 256) {
        const float v = e[i];
        mn = fminf(mn, v); mx = fmaxf(mx, v);
    }
    smin[threadIdx.x] = mn; smax[threadIdx.x] = mx;
    __syncthreads();
    for (int s = 128; s > 0; s >>= 1) {
        if (threadIdx.x < s) {
            smin[threadIdx.x] = fminf(smin[threadIdx.x], smin[threadIdx.x + s]);
            smax[threadIdx.x] = fmaxf(smax[threadIdx.x], smax[threadIdx.x + s]);
        }
        __syncthreads();
    }
    if (threadIdx.x == 0) { emin[b] = smin[0]; emax[b] = smax[0]; }
}

// ---------------- K5: threshold + degenerate one-hot + OR over boxes ----------------
// Each thread produces 4 consecutive pixels so the mask store is a b128.
__global__ void k5_mask(const float* __restrict__ energy,
                        const float* __restrict__ emin, const float* __restrict__ emax,
                        const int* __restrict__ lx, const int* __restrict__ ly,
                        const int* __restrict__ rx, const int* __restrict__ ry,
                        const int* __restrict__ degen, float* __restrict__ mask) {
    const int gid = blockIdx.x * 256 + threadIdx.x;   // 0..8191 (groups of 4 pixels)
    const int n = gid >> 10;                          // / (4096/4)
    const int p0 = (gid & 1023) * 4;                  // first pixel of the 4
    float4 m = make_float4(0.f, 0.f, 0.f, 0.f);
    float* mp = &m.x;
    for (int k = 0; k < KBOX; ++k) {
        const int b = n * KBOX + k;
        const int dg = degen[b];
        const float mn = emin[b];
        const float mxInv = 1.0f / emax[b];
        const int oneHot = ly[b] * WW + lx[b];
        #pragma unroll
        for (int j = 0; j < 4; ++j) {
            const int pixel = p0 + j;
            bool att;
            if (dg) {
                att = (pixel == oneHot);
            } else {
                const float ev = energy[(size_t)b * HW + pixel];
                att = ((ev - mn) * mxInv) > 0.6f;
            }
            if (att) mp[j] = 1.0f;
        }
    }
    *(float4*)(mask + (size_t)n * HW + p0) = m;
}

// ---------------- K6: main HBM pass: diff^2 split by mask, per-block partials ----
// float4 streams for S, T, and mask: one b128 load per 16 bytes to saturate HBM.
__global__ void k6_reduce(const float4* __restrict__ S4, const float4* __restrict__ T4,
                          const float* __restrict__ mask, float* __restrict__ part) {
    __shared__ float sg[256], st[256];
    const size_t total4 = ((size_t)NIMG * NCH * HW) / 4;   // 2097152
    float g = 0.0f, s = 0.0f;
    for (size_t i = (size_t)blockIdx.x * 256 + threadIdx.x; i < total4;
         i += (size_t)RED_BLOCKS * 256) {
        const int n    = (int)(i >> 18);        // element (4i) >> 20
        const int pix4 = (int)(i & 1023);       // (4i & 4095) / 4
        const float4 tv = T4[i];
        const float4 sv = S4[i];
        const float4 mv = *(const float4*)(mask + (size_t)n * HW + pix4 * 4);
        float d;
        d = tv.x - sv.x; d *= d; s += d; g += d * mv.x;
        d = tv.y - sv.y; d *= d; s += d; g += d * mv.y;
        d = tv.z - sv.z; d *= d; s += d; g += d * mv.z;
        d = tv.w - sv.w; d *= d; s += d; g += d * mv.w;
    }
    sg[threadIdx.x] = g; st[threadIdx.x] = s;
    __syncthreads();
    for (int k = 128; k > 0; k >>= 1) {
        if (threadIdx.x < k) {
            sg[threadIdx.x] += sg[threadIdx.x + k];
            st[threadIdx.x] += st[threadIdx.x + k];
        }
        __syncthreads();
    }
    if (threadIdx.x == 0) {
        part[2 * blockIdx.x]     = sg[0];
        part[2 * blockIdx.x + 1] = st[0];
    }
}

// ---------------- K7: final deterministic reduce -> scalar loss ----------------
__global__ void k7_final(const float* __restrict__ part, float* __restrict__ out) {
    __shared__ float sg[256], st[256];
    float g = 0.0f, s = 0.0f;
    for (int j = threadIdx.x; j < RED_BLOCKS; j += 256) {
        g += part[2 * j];
        s += part[2 * j + 1];
    }
    sg[threadIdx.x] = g; st[threadIdx.x] = s;
    __syncthreads();
    for (int k = 128; k > 0; k >>= 1) {
        if (threadIdx.x < k) {
            sg[threadIdx.x] += sg[threadIdx.x + k];
            st[threadIdx.x] += st[threadIdx.x + k];
        }
        __syncthreads();
    }
    if (threadIdx.x == 0) {
        const float G = sg[0], Tt = st[0];
        out[0] = sqrtf(G + 1e-8f) + sqrtf((Tt - G) + 1e-8f);
    }
}

extern "C" void kernel_launch(void* const* d_in, const int* in_sizes, int n_in,
                              void* d_out, int out_size, void* d_ws, size_t ws_size,
                              hipStream_t stream) {
    const float* S  = (const float*)d_in[0];   // preds_S (8,256,64,64)
    const float* T  = (const float*)d_in[1];   // preds_T (8,256,64,64)
    const float* BB = (const float*)d_in[2];   // gt_bboxes (8,16,4)
    float* out = (float*)d_out;

    char* ws = (char*)d_ws;
    int*       lx     = (int*)(ws + OFF_LX);
    int*       ly     = (int*)(ws + OFF_LY);
    int*       rx     = (int*)(ws + OFF_RX);
    int*       ry     = (int*)(ws + OFF_RY);
    float*     cnt    = (float*)(ws + OFF_CNT);
    int*       degen  = (int*)(ws + OFF_DEG);
    _Float16*  fgt    = (_Float16*)(ws + OFF_FGT);
    float*     energy = (float*)(ws + OFF_ENERGY);
    float*     emin   = (float*)(ws + OFF_EMIN);
    float*     emax   = (float*)(ws + OFF_EMAX);
    float*     mask   = (float*)(ws + OFF_MASK);
    float*     part   = (float*)(ws + OFF_PART);

    k1_boxes      <<<1,          128, 0, stream>>>(BB, lx, ly, rx, ry, cnt, degen);
    k2_featgt     <<<NBOX,       256, 0, stream>>>(T, lx, ly, rx, ry, cnt, fgt);
    k3_energy_wmma<<<NIMG * 32,  256, 0, stream>>>(T, fgt, energy);
    k4_minmax     <<<NBOX,       256, 0, stream>>>(energy, emin, emax);
    k5_mask       <<<32,         256, 0, stream>>>(energy, emin, emax, lx, ly, rx, ry, degen, mask);
    k6_reduce     <<<RED_BLOCKS, 256, 0, stream>>>((const float4*)S, (const float4*)T, mask, part);
    k7_final      <<<1,          256, 0, stream>>>(part, out);
}
